// GraphAttentionLayer_171798692626
// MI455X (gfx1250) — matrix operand order repaired and checked by reading, not compiled, not run on hardware
//
#include <hip/hip_runtime.h>

// ---------------------------------------------------------------------------
// GAT layer, fused for MI455X (gfx1250, wave32, WMMA).
//   N = 8192, F_in = 512, F_out = 128
// k1: Wh = h @ W         fp32 WMMA 16x16x4  (+ bf16 transposed copy WhT)
// k2: Wh1/Wh2 row dots   wave-per-row, __shfl_xor reduce
// k3: fused masked softmax + (attention @ Wh) via bf16 WMMA 16x16x32 + ELU
//     (j-tile = 128 -> 4 WMMAs per barrier pair)
// ---------------------------------------------------------------------------

typedef float  f32x2   __attribute__((ext_vector_type(2)));
typedef float  f32x8   __attribute__((ext_vector_type(8)));
typedef __bf16 bf16x16 __attribute__((ext_vector_type(16)));
typedef unsigned int u32x4 __attribute__((ext_vector_type(4)));

#define N_NODES 8192
#define F_IN    512
#define F_OUT   128
#define ALPHA   0.2f
#define NEG_BIG (-9e15f)

// round-to-nearest-even float -> bf16 bits
__device__ __forceinline__ unsigned short f2bf(float f) {
  unsigned int u = __builtin_bit_cast(unsigned int, f);
  u += 0x7FFFu + ((u >> 16) & 1u);
  return (unsigned short)(u >> 16);
}

// ---------------------------------------------------------------------------
// Kernel 1: Wh = h @ W with fp32 WMMA (exact fp32 for the score path).
// Block = 256 thr (8 waves). Block b -> rows [16b,16b+16); wave w -> cols
// [16w,16w+16). K loop over 512 in steps of 4 (V_WMMA_F32_16X16X4_F32).
// Writes Wh fp32 [N,128] and WhT bf16 [128,N] (B operand layout for k3).
// ---------------------------------------------------------------------------
__global__ __launch_bounds__(256) void gat_wh_gemm(
    const float* __restrict__ h, const float* __restrict__ W,
    float* __restrict__ Wh, unsigned short* __restrict__ WhT) {
  const int r0   = blockIdx.x * 16;
  const int w    = threadIdx.x >> 5;
  const int lane = threadIdx.x & 31;
  const int colL = lane & 15;
  const int hi   = lane >> 4;
  const int colG = w * 16 + colL;
  const int arow = r0 + colL;              // A-matrix row for this lane (M = lane&15)

  const float* hrow = h + (size_t)arow * F_IN;
  f32x8 c = {};
  for (int k = 0; k < F_IN; k += 4) {
    const int kk = k + (hi << 1);          // lanes 0-15: K=k,k+1 ; lanes 16-31: K=k+2,k+3
    f32x2 A = *(const f32x2*)(hrow + kk);
    f32x2 B;
    B.x = W[kk * F_OUT + colG];
    B.y = W[(kk + 1) * F_OUT + colG];
    c = __builtin_amdgcn_wmma_f32_16x16x4_f32(false, A, false, B,
                                              (short)0, c, false, false);
  }
#pragma unroll
  for (int g = 0; g < 8; ++g) {            // C: VGPR g -> row g (+8 for hi lanes)
    const int r = r0 + g + hi * 8;
    const float v = c[g];
    Wh[(size_t)r * F_OUT + colG] = v;
    WhT[(size_t)colG * N_NODES + r] = f2bf(v);
  }
}

// ---------------------------------------------------------------------------
// Kernel 2: Wh1[i] = Wh[i,:]@a[:128], Wh2[i] = Wh[i,:]@a[128:].
// One wave per row; lane loads float4, 5-step __shfl_xor tree (wave32).
// ---------------------------------------------------------------------------
__global__ __launch_bounds__(256) void gat_rowdots(
    const float* __restrict__ Wh, const float* __restrict__ a,
    float* __restrict__ Wh1, float* __restrict__ Wh2) {
  const int row  = blockIdx.x * 8 + (threadIdx.x >> 5);
  const int lane = threadIdx.x & 31;
  const float4 v  = *(const float4*)(Wh + (size_t)row * F_OUT + lane * 4);
  const float4 a1 = *(const float4*)(a + lane * 4);
  const float4 a2 = *(const float4*)(a + F_OUT + lane * 4);
  float d1 = v.x * a1.x + v.y * a1.y + v.z * a1.z + v.w * a1.w;
  float d2 = v.x * a2.x + v.y * a2.y + v.z * a2.z + v.w * a2.w;
#pragma unroll
  for (int off = 16; off > 0; off >>= 1) {
    d1 += __shfl_xor(d1, off, 32);
    d2 += __shfl_xor(d2, off, 32);
  }
  if (lane == 0) { Wh1[row] = d1; Wh2[row] = d2; }
}

// ---------------------------------------------------------------------------
// Kernel 3: fused attention. Block owns 16 rows. Streams its 512 KB adj slab
// (L2-resident across passes): pass1a row max, pass1b row sum-of-exp, pass2
// 64 j-tiles of 128: build bf16 P tile in LDS (one packed ds_store_b128 per
// thread), each of 8 waves issues 4 V_WMMA_F32_16X16X32_BF16 per barrier
// pair into its 16x16 f32 accumulator. Normalize + ELU out.
// ---------------------------------------------------------------------------
__global__ __launch_bounds__(256) void gat_attn(
    const int* __restrict__ adj, const float* __restrict__ Wh1,
    const float* __restrict__ Wh2, const unsigned short* __restrict__ WhT,
    float* __restrict__ out) {
  __shared__ float wh2s[N_NODES];                        // 32 KB stage of Wh2
  __shared__ float wh1s[16];
  __shared__ float red[256];
  __shared__ float ms[16];
  __shared__ float isum[16];
  __shared__ __align__(16) unsigned short Pt[16 * 128];  // 4 KB bf16 P tile

  const int t  = threadIdx.x;
  const int r0 = blockIdx.x * 16;

  for (int j = t; j < N_NODES; j += 256) wh2s[j] = Wh2[j];
  if (t < 16) wh1s[t] = Wh1[r0 + t];
  __syncthreads();

  const int prow = t >> 4;                 // this thread's row (0..15)
  const int tj   = t & 15;
  const int* arow = adj + (size_t)(r0 + prow) * N_NODES;
  const float w1 = wh1s[prow];

  // ---- pass 1a: row max of masked leaky scores ----
  float m = NEG_BIG;
  for (int i = 0; i < 128; ++i) {
    const int j = i * 64 + tj * 4;
    const int4 av = *(const int4*)(arow + j);
#pragma unroll
    for (int q = 0; q < 4; ++q) {
      const int aq = (q == 0) ? av.x : (q == 1) ? av.y : (q == 2) ? av.z : av.w;
      float e = w1 + wh2s[j + q];
      e = (e >= 0.f) ? e : ALPHA * e;
      if (aq > 0) m = fmaxf(m, e);
    }
  }
  red[t] = m;
  __syncthreads();
  if (t < 16) {
    float mm = red[t * 16];
    for (int k = 1; k < 16; ++k) mm = fmaxf(mm, red[t * 16 + k]);
    ms[t] = mm;
  }
  __syncthreads();
  const float mrow = ms[prow];

  // ---- pass 1b: row sum of exp(score - max) (masked -> exp(-big) == 0;
  //      all-masked row -> max==NEG_BIG, every term exp(0)=1 => uniform) ----
  float ssum = 0.f;
  for (int i = 0; i < 128; ++i) {
    const int j = i * 64 + tj * 4;
    const int4 av = *(const int4*)(arow + j);
#pragma unroll
    for (int q = 0; q < 4; ++q) {
      const int aq = (q == 0) ? av.x : (q == 1) ? av.y : (q == 2) ? av.z : av.w;
      float e = w1 + wh2s[j + q];
      e = (e >= 0.f) ? e : ALPHA * e;
      const float sc = (aq > 0) ? e : NEG_BIG;
      ssum += __expf(sc - mrow);
    }
  }
  red[t] = ssum;
  __syncthreads();
  if (t < 16) {
    float s = 0.f;
    for (int k = 0; k < 16; ++k) s += red[t * 16 + k];
    isum[t] = 1.f / s;
  }
  __syncthreads();

  // ---- pass 2: acc += P(16x128 bf16) @ Wh(128x16 bf16), f32 accumulate ----
  const int w    = t >> 5;
  const int lane = t & 31;
  const int colL = lane & 15;
  const int hi   = lane >> 4;
  const int k0   = hi * 8;                       // A: K slice base per lane-half
  const unsigned short* bbase =
      WhT + (size_t)(w * 16 + colL) * N_NODES + hi * 8;   // B: K slice base per lane-half
  const int jj = tj * 8;                         // 8 probabilities per thread/tile
  f32x8 acc = {};

  for (int tile = 0; tile < 64; ++tile) {
    const int j0 = tile * 128;
    // build bf16 P tile: 8 masked probabilities, packed -> one b128 LDS store
    const int4 av0 = *(const int4*)(arow + j0 + jj);
    const int4 av1 = *(const int4*)(arow + j0 + jj + 4);
    u32x4 pk;
#pragma unroll
    for (int q = 0; q < 8; ++q) {
      const int aq = (q < 4)
          ? ((q == 0) ? av0.x : (q == 1) ? av0.y : (q == 2) ? av0.z : av0.w)
          : ((q == 4) ? av1.x : (q == 5) ? av1.y : (q == 6) ? av1.z : av1.w);
      float e = w1 + wh2s[j0 + jj + q];
      e = (e >= 0.f) ? e : ALPHA * e;
      const float p = (aq > 0) ? __expf(e - mrow) : 0.f;
      const unsigned int b = f2bf(p);
      if (q & 1) pk[q >> 1] |= b << 16;
      else       pk[q >> 1]  = b;
    }
    *(u32x4*)(Pt + prow * 128 + jj) = pk;
    __syncthreads();

    // 4 back-to-back WMMAs over K = 128
#pragma unroll
    for (int kk = 0; kk < 4; ++kk) {
      union { u32x4 q[2]; bf16x16 v; } A, B;
      const unsigned short* pa = Pt + colL * 128 + kk * 32 + k0;
      A.q[0] = *(const u32x4*)(pa);                     // K = k0..k0+7
      A.q[1] = *(const u32x4*)(pa + 16);                // K = k0+16..k0+23
      const unsigned short* bp = bbase + j0 + kk * 32;
      B.q[0] = *(const u32x4*)(bp);                     // K = k0..k0+7
      B.q[1] = *(const u32x4*)(bp + 16);                // K = k0+16..k0+23
      acc = __builtin_amdgcn_wmma_f32_16x16x32_bf16(false, A.v, false, B.v,
                                                    (short)0, acc, false, false);
    }
    __syncthreads();
  }

  // ---- normalize + ELU + store ----
  const int colG = w * 16 + colL;
#pragma unroll
  for (int g = 0; g < 8; ++g) {
    const int r = g + hi * 8;
    float x = acc[g] * isum[r];
    x = (x > 0.f) ? x : (__expf(x) - 1.f);
    out[(size_t)(r0 + r) * F_OUT + colG] = x;
  }
}

// ---------------------------------------------------------------------------
extern "C" void kernel_launch(void* const* d_in, const int* in_sizes, int n_in,
                              void* d_out, int out_size, void* d_ws,
                              size_t ws_size, hipStream_t stream) {
  const float* h   = (const float*)d_in[0];   // [8192,512] f32
  const int*   adj = (const int*)d_in[1];     // [8192,8192] i32
  const float* W   = (const float*)d_in[2];   // [512,128] f32
  const float* a   = (const float*)d_in[3];   // [256,1] f32
  float* out = (float*)d_out;                 // [8192,128] f32

  char* ws = (char*)d_ws;                     // ~6.1 MB used
  float*          Wh  = (float*)ws;                                  // 4 MB
  unsigned short* WhT = (unsigned short*)(ws + (4u << 20));          // 2 MB
  float*          Wh1 = (float*)(ws + (6u << 20));                   // 32 KB
  float*          Wh2 = Wh1 + N_NODES;                               // 32 KB

  gat_wh_gemm<<<N_NODES / 16, 256, 0, stream>>>(h, W, Wh, WhT);
  gat_rowdots<<<N_NODES / 8, 256, 0, stream>>>(Wh, a, Wh1, Wh2);
  gat_attn<<<N_NODES / 16, 256, 0, stream>>>(adj, Wh1, Wh2, WhT, out);
}
